// MultiHeadDelta_3204045603562
// MI455X (gfx1250) — compile-verified, matrix operand order checked by reading, not run on hardware
//
#include <hip/hip_runtime.h>

namespace {

constexpr int Bn  = 16;
constexpr int Ln  = 2048;
constexpr int Hn  = 512;
constexpr int NHn = 8;
constexpr int HHn = 64;
constexpr int Vn  = 32000;

typedef __attribute__((ext_vector_type(8)))  __bf16 v8bf;
typedef __attribute__((ext_vector_type(16))) __bf16 v16bf;
typedef __attribute__((ext_vector_type(8)))  float  v8f;

__device__ __forceinline__ unsigned short f2bf(float x) {
  unsigned int u = __float_as_uint(x);
  u += 0x7FFFu + ((u >> 16) & 1u);   // round-to-nearest-even
  return (unsigned short)(u >> 16);
}

// Load one bf16 fragment (A 16x32, or B as rows of Wt) for this lane:
// caller passes p = base_row*K + 8*(lane>>4); halves are [p,p+8) and [p+16,p+24).
__device__ __forceinline__ v16bf ld_frag(const unsigned short* p) {
  v8bf lo = *(const v8bf*)(p);
  v8bf hi = *(const v8bf*)(p + 16);
  v16bf r;
#pragma unroll
  for (int i = 0; i < 8; ++i) { r[i] = lo[i]; r[i + 8] = hi[i]; }
  return r;
}

// ---------------- embedding gather + bf16 convert -------------------------
__global__ __launch_bounds__(128) void gather_embed_kernel(
    const int* __restrict__ seq, const float* __restrict__ embed,
    unsigned short* __restrict__ e16) {
  int row = blockIdx.x;
  int tok = seq[row];
  float4 v = ((const float4*)(embed + (size_t)tok * Hn))[threadIdx.x];
  ushort4 o;
  o.x = f2bf(v.x); o.y = f2bf(v.y); o.z = f2bf(v.z); o.w = f2bf(v.w);
  ((ushort4*)(e16 + (size_t)row * Hn))[threadIdx.x] = o;
}

// ---------------- fp32 (R,C) -> bf16 (C,R) tiled transpose ----------------
__global__ __launch_bounds__(256) void transpose_cvt_kernel(
    const float* __restrict__ in, unsigned short* __restrict__ out,
    int R, int C) {
  __shared__ float tile[32][33];
  int tx = threadIdx.x & 31, ty = threadIdx.x >> 5;
  int bc = blockIdx.x * 32, br = blockIdx.y * 32;
#pragma unroll
  for (int i = ty; i < 32; i += 8) {
    int rr = br + i, cc = bc + tx;
    tile[i][tx] = (rr < R && cc < C) ? in[(size_t)rr * C + cc] : 0.f;
  }
  __syncthreads();
#pragma unroll
  for (int i = ty; i < 32; i += 8) {
    int orow = bc + i, ocol = br + tx;
    if (orow < C && ocol < R) out[(size_t)orow * R + ocol] = f2bf(tile[tx][i]);
  }
}

// ---------------- LayerNorm (H=512), fp32 in -> bf16 out ------------------
__global__ __launch_bounds__(128) void layernorm_kernel(
    const float* __restrict__ x, const float* __restrict__ g,
    const float* __restrict__ bta, unsigned short* __restrict__ h16) {
  int row = blockIdx.x;
  float4 v = ((const float4*)(x + (size_t)row * Hn))[threadIdx.x];
  float s  = v.x + v.y + v.z + v.w;
  float sq = v.x * v.x + v.y * v.y + v.z * v.z + v.w * v.w;
#pragma unroll
  for (int off = 16; off >= 1; off >>= 1) {
    s  += __shfl_xor(s, off, 32);
    sq += __shfl_xor(sq, off, 32);
  }
  __shared__ float red[2][4];
  int wv = threadIdx.x >> 5, ln = threadIdx.x & 31;
  if (ln == 0) { red[0][wv] = s; red[1][wv] = sq; }
  __syncthreads();
  s  = red[0][0] + red[0][1] + red[0][2] + red[0][3];
  sq = red[1][0] + red[1][1] + red[1][2] + red[1][3];
  float mu  = s * (1.f / Hn);
  float var = sq * (1.f / Hn) - mu * mu;
  float rs  = rsqrtf(var + 1e-5f);
  float4 gg = ((const float4*)g)[threadIdx.x];
  float4 bb = ((const float4*)bta)[threadIdx.x];
  ushort4 o;
  o.x = f2bf((v.x - mu) * rs * gg.x + bb.x);
  o.y = f2bf((v.y - mu) * rs * gg.y + bb.y);
  o.z = f2bf((v.z - mu) * rs * gg.z + bb.z);
  o.w = f2bf((v.w - mu) * rs * gg.w + bb.w);
  ((ushort4*)(h16 + (size_t)row * Hn))[threadIdx.x] = o;
}

// ---------------- pack h_all[:, L-1] rows for the q projection ------------
__global__ __launch_bounds__(256) void gather_last_kernel(
    const unsigned short* __restrict__ h16, unsigned short* __restrict__ hq) {
  int b = blockIdx.x;
  const unsigned short* src = h16 + ((size_t)b * Ln + (Ln - 1)) * Hn;
  for (int i = threadIdx.x; i < Hn; i += blockDim.x) hq[b * Hn + i] = src[i];
}

// ---------------- WMMA bf16 GEMM: out = A(M,K) @ Wt(N,K)^T + bias ---------
// Each wave owns a 16(M) x 64(N) strip (4 accumulators, A fragment reused
// 4x). K-loop is ping-pong software-pipelined (unroll 2): two fragment sets
// with fixed register roles are alternately consumed and refilled in place,
// so waits stay partial and loads overlap the WMMA pipe. Optional ReLU /
// bf16 store / fused embedding residual (re-gathered via seq).
template <int WM, int WN, bool RELU, bool OUTBF16, bool RESID>
__global__ __launch_bounds__(WM* WN * 32) void wmma_gemm_kernel(
    const unsigned short* __restrict__ A, const unsigned short* __restrict__ Bw,
    const float* __restrict__ bias, void* __restrict__ out,
    int M, int N, int K,
    const int* __restrict__ seq, const float* __restrict__ embed) {
  int wave = threadIdx.x >> 5;
  int lane = threadIdx.x & 31;
  int wm = wave % WM;
  int wn = wave / WM;
  int m0 = (blockIdx.x * WM + wm) * 16;
  int n0 = (blockIdx.y * WN + wn) * 64;
  int r  = lane & 15, hh = lane >> 4;

  int arow = m0 + r; if (arow > M - 1) arow = M - 1;
  const unsigned short* ap = A + (size_t)arow * K + 8 * hh;
  const unsigned short* bp[4];
#pragma unroll
  for (int t = 0; t < 4; ++t) {
    int brow = n0 + 16 * t + r; if (brow > N - 1) brow = N - 1;
    bp[t] = Bw + (size_t)brow * K + 8 * hh;
  }

  v8f acc[4] = {};
  // prologue: K-steps 0 and 32 into the two fragment sets (K % 64 == 0)
  v16bf a0 = ld_frag(ap);
  v16bf b0[4];
#pragma unroll
  for (int t = 0; t < 4; ++t) b0[t] = ld_frag(bp[t]);
  v16bf a1 = ld_frag(ap + 32);
  v16bf b1[4];
#pragma unroll
  for (int t = 0; t < 4; ++t) b1[t] = ld_frag(bp[t] + 32);

  for (int k0 = 64; k0 < K; k0 += 64) {
    // consume even-step set, refill it with step k0
#pragma unroll
    for (int t = 0; t < 4; ++t)
      acc[t] = __builtin_amdgcn_wmma_f32_16x16x32_bf16(
          false, a0, false, b0[t], (short)0, acc[t], false, false);
    a0 = ld_frag(ap + k0);
#pragma unroll
    for (int t = 0; t < 4; ++t) b0[t] = ld_frag(bp[t] + k0);
    // consume odd-step set, refill it with step k0+32
#pragma unroll
    for (int t = 0; t < 4; ++t)
      acc[t] = __builtin_amdgcn_wmma_f32_16x16x32_bf16(
          false, a1, false, b1[t], (short)0, acc[t], false, false);
    a1 = ld_frag(ap + k0 + 32);
#pragma unroll
    for (int t = 0; t < 4; ++t) b1[t] = ld_frag(bp[t] + k0 + 32);
  }
  // epilogue: last two K-steps
#pragma unroll
  for (int t = 0; t < 4; ++t)
    acc[t] = __builtin_amdgcn_wmma_f32_16x16x32_bf16(
        false, a0, false, b0[t], (short)0, acc[t], false, false);
#pragma unroll
  for (int t = 0; t < 4; ++t)
    acc[t] = __builtin_amdgcn_wmma_f32_16x16x32_bf16(
        false, a1, false, b1[t], (short)0, acc[t], false, false);

#pragma unroll
  for (int t = 0; t < 4; ++t) {
    int col = n0 + 16 * t + r;
    if (col >= N) continue;
    float bv = bias[col];
#pragma unroll
    for (int i = 0; i < 8; ++i) {
      int row = m0 + i + 8 * hh;
      if (row >= M) continue;
      float v = acc[t][i] + bv;
      if (RESID) v += embed[(size_t)seq[row] * Hn + col];
      if (RELU)  v = fmaxf(v, 0.f);
      if (OUTBF16) ((unsigned short*)out)[(size_t)row * N + col] = f2bf(v);
      else         ((float*)out)[(size_t)row * N + col] = v;
    }
  }
}

// ---------------- Tensor Data Mover path for the scan ---------------------
#if defined(__gfx1250__) && __has_builtin(__builtin_amdgcn_tensor_load_to_lds)
#define DELTA_USE_TDM 1
#else
#define DELTA_USE_TDM 0
#endif

// Compile-time probe: report in stderr which scan data-path was selected.
#if DELTA_USE_TDM
#if __clang_major__ >= 23
#warning "CDNA5 probe: delta_scan uses TDM tensor_load_to_lds (6-arg, clang>=23)"
#else
#warning "CDNA5 probe: delta_scan uses TDM tensor_load_to_lds (5-arg, clang<23)"
#endif
#elif defined(__gfx1250__)
#warning "CDNA5 probe: tensor_load_to_lds builtin NOT available; delta_scan uses lane-load fallback"
#endif

#if DELTA_USE_TDM
typedef __attribute__((ext_vector_type(4))) unsigned int u32x4;
typedef __attribute__((ext_vector_type(8))) int          i32x8;
typedef __attribute__((ext_vector_type(4))) int          i32x4;

// One TDM 3D tile: X = 64 floats (head slice), Y = 2 segments (k at +0,
// v at +512 floats), Z = 8 sequence steps (row stride 1024 floats).
// LDS result is linear: [step][k|v][64 floats]  == float4[8][32].
__device__ __forceinline__ void tdm_load_group(unsigned long long ga,
                                               unsigned lds_addr) {
  u32x4 g0;
  g0[0] = 1u;                                                  // count=1
  g0[1] = lds_addr;                                            // LDS byte addr
  g0[2] = (unsigned)ga;                                        // global[31:0]
  g0[3] = (unsigned)((ga >> 32) & 0x01FFFFFFu) | (2u << 30);   // global[56:32] | type=2
  i32x8 g1;
  g1[0] = (int)(2u << 16);          // data_size = 4B
  g1[1] = (int)(64u << 16);         // tensor_dim0 = 64
  g1[2] = (int)(2u << 16);          // tensor_dim1 = 2
  g1[3] = (int)(64u << 16);         // tile_dim0 = 64
  g1[4] = (int)(2u | (8u << 16));   // tile_dim1 = 2, tile_dim2 = 8
  g1[5] = 512;                      // tensor_dim0_stride = 512 elems (k->v)
  g1[6] = (int)(1024u << 16);       // tensor_dim1_stride = 1024 elems (row)
  g1[7] = 0;
  i32x4 g2;
  g2[0] = 8;                        // tensor_dim2 = 8
  g2[1] = 0; g2[2] = 0; g2[3] = 0;
  i32x4 g3;
  g3[0] = 0; g3[1] = 0; g3[2] = 0; g3[3] = 0;
#if __clang_major__ >= 23
  i32x8 gpad = {};
  __builtin_amdgcn_tensor_load_to_lds(g0, g1, g2, g3, gpad, 0);
#else
  __builtin_amdgcn_tensor_load_to_lds(g0, g1, g2, g3, 0);
#endif
}
#endif  // DELTA_USE_TDM

// ---------------- sequential delta-rule scan ------------------------------
// One block per (b, head). M (64x64 fp32) in registers: thread t owns row
// i=t>>2, cols [16g,16g+16) with g=t&3. k/v streamed in 8-step groups:
// TDM DMA into double-buffered LDS (tensorcnt-synced) when available, else
// 256 coalesced lane loads. Finishes with ctx = M @ q stored as bf16.
__global__ __launch_bounds__(256) void delta_scan_kernel(
    const float* __restrict__ kv, const float* __restrict__ q,
    unsigned short* __restrict__ ctx) {
  int b = blockIdx.x >> 3;
  int n = blockIdx.x & 7;
  int t = threadIdx.x;
  int i = t >> 2;
  int g = t & 3;
  float m[16];
#pragma unroll
  for (int j = 0; j < 16; ++j) m[j] = 0.f;

  const float* base = kv + (size_t)b * Ln * (2 * Hn) + n * HHn;
  const int steps = Ln - 1;  // 2047
  __shared__ float4 sq_tail[16];

#if DELTA_USE_TDM
  __shared__ float4 skv[2][8][32];
  unsigned long long gbase = (unsigned long long)(size_t)base;
  if (t < 32) tdm_load_group(gbase, (unsigned)(size_t)&skv[0][0][0]);
  int bi = 0;
  for (int l0 = 0; l0 < steps; l0 += 8, bi ^= 1) {
    __syncthreads();  // everyone done with the buffer we are about to refill
    if (t < 32) {
      if (l0 + 8 < steps) {
        tdm_load_group(gbase + (unsigned long long)(l0 + 8) * (2 * Hn * 4),
                       (unsigned)(size_t)&skv[bi ^ 1][0][0]);
        __builtin_amdgcn_s_wait_tensorcnt(1);  // group l0 landed in LDS
      } else {
        __builtin_amdgcn_s_wait_tensorcnt(0);
      }
    }
    __syncthreads();  // publish group l0
    const float4(*sg)[32] = skv[bi];
    int lim = steps - l0; if (lim > 8) lim = 8;
    for (int u = 0; u < lim; ++u) {
      const float4* kk = &sg[u][g * 4];
      float4 a0 = kk[0], a1 = kk[1], a2 = kk[2], a3 = kk[3];
      float kj[16] = {a0.x, a0.y, a0.z, a0.w, a1.x, a1.y, a1.z, a1.w,
                      a2.x, a2.y, a2.z, a2.w, a3.x, a3.y, a3.z, a3.w};
      float part = 0.f, k2s = 0.f;
#pragma unroll
      for (int j = 0; j < 16; ++j) { part += m[j] * kj[j]; k2s += kj[j] * kj[j]; }
      part += __shfl_xor(part, 1, 32); part += __shfl_xor(part, 2, 32);
      k2s  += __shfl_xor(k2s, 1, 32);  k2s  += __shfl_xor(k2s, 2, 32);
      float vi = ((const float*)&sg[u][16])[i];
      float delta = vi - part / (k2s + 1e-6f);
#pragma unroll
      for (int j = 0; j < 16; ++j) m[j] = fmaf(delta, kj[j], m[j]);
    }
  }
#else
  __shared__ float4 skv[8][32];  // per step: float4[0..15]=k, [16..31]=v
  int slot = t >> 5;             // which step of the 8-group this thread loads
  int id   = t & 31;             // float4 index within the step
  for (int l0 = 0; l0 < steps; l0 += 8) {
    int ls = l0 + slot; if (ls > steps - 1) ls = steps - 1;
    const float* rb = base + (size_t)ls * (2 * Hn);
    const float* addr = (id < 16) ? (rb + 4 * id) : (rb + Hn + 4 * (id - 16));
    float4 val = *(const float4*)addr;
    int lp = l0 + 8 + slot; if (lp > steps - 1) lp = steps - 1;
    const float* pb = base + (size_t)lp * (2 * Hn);
    __builtin_prefetch((id < 16) ? (pb + 4 * id) : (pb + Hn + 4 * (id - 16)), 0, 1);
    __syncthreads();
    skv[slot][id] = val;
    __syncthreads();
    int lim = steps - l0; if (lim > 8) lim = 8;
    for (int u = 0; u < lim; ++u) {
      const float4* kk = &skv[u][g * 4];
      float4 a0 = kk[0], a1 = kk[1], a2 = kk[2], a3 = kk[3];
      float kj[16] = {a0.x, a0.y, a0.z, a0.w, a1.x, a1.y, a1.z, a1.w,
                      a2.x, a2.y, a2.z, a2.w, a3.x, a3.y, a3.z, a3.w};
      float part = 0.f, k2s = 0.f;
#pragma unroll
      for (int j = 0; j < 16; ++j) { part += m[j] * kj[j]; k2s += kj[j] * kj[j]; }
      part += __shfl_xor(part, 1, 32); part += __shfl_xor(part, 2, 32);
      k2s  += __shfl_xor(k2s, 1, 32);  k2s  += __shfl_xor(k2s, 2, 32);
      float vi = ((const float*)&skv[u][16])[i];
      float delta = vi - part / (k2s + 1e-6f);
#pragma unroll
      for (int j = 0; j < 16; ++j) m[j] = fmaf(delta, kj[j], m[j]);
    }
  }
#endif

  // ctx = M @ q
  __syncthreads();
  if (t < 16) sq_tail[t] = *(const float4*)(q + (size_t)b * Hn + n * HHn + 4 * t);
  __syncthreads();
  const float4* qq = &sq_tail[g * 4];
  float4 q0 = qq[0], q1 = qq[1], q2 = qq[2], q3 = qq[3];
  float qj[16] = {q0.x, q0.y, q0.z, q0.w, q1.x, q1.y, q1.z, q1.w,
                  q2.x, q2.y, q2.z, q2.w, q3.x, q3.y, q3.z, q3.w};
  float part = 0.f;
#pragma unroll
  for (int j = 0; j < 16; ++j) part += m[j] * qj[j];
  part += __shfl_xor(part, 1, 32); part += __shfl_xor(part, 2, 32);
  if (g == 0) ctx[(size_t)b * Hn + n * HHn + i] = f2bf(part);
}

}  // namespace

extern "C" void kernel_launch(void* const* d_in, const int* in_sizes, int n_in,
                              void* d_out, int out_size, void* d_ws, size_t ws_size,
                              hipStream_t stream) {
  (void)in_sizes; (void)n_in; (void)out_size; (void)ws_size;
  const int*   seq   = (const int*)d_in[0];
  const float* embed = (const float*)d_in[1];
  const float* ff_w1 = (const float*)d_in[2];
  const float* ff_b1 = (const float*)d_in[3];
  const float* ff_w2 = (const float*)d_in[4];
  const float* ff_b2 = (const float*)d_in[5];
  const float* ln_g  = (const float*)d_in[6];
  const float* ln_b  = (const float*)d_in[7];
  const float* wk    = (const float*)d_in[8];
  const float* bk    = (const float*)d_in[9];
  const float* wv    = (const float*)d_in[10];
  const float* bv    = (const float*)d_in[11];
  const float* wq    = (const float*)d_in[12];
  const float* bq    = (const float*)d_in[13];
  const float* w_op  = (const float*)d_in[14];
  const float* b_op  = (const float*)d_in[15];
  const float* w_out = (const float*)d_in[16];
  const float* b_out = (const float*)d_in[17];
  float* out = (float*)d_out;

  // ---- workspace layout (aliased pool reuses dead buffers) ----
  char* ws = (char*)d_ws;
  const size_t MB = 1024 * 1024;
  unsigned short* e16 = (unsigned short*)ws;             // 32 MiB [dies after GEMM1]
  float*          x32 = (float*)ws;                      // 64 MiB [aliases e16]
  unsigned short* h1  = (unsigned short*)(ws + 64 * MB); // 64 MiB
  float*          kv  = (float*)ws;                      // 128 MiB [aliases all above]
  size_t off = 128 * MB;
  auto take = [&](size_t bytes) -> char* {
    char* p = ws + off; off += (bytes + 255) & ~(size_t)255; return p;
  };
  unsigned short* h16   = (unsigned short*)take((size_t)Bn * Ln * Hn * 2);
  unsigned short* w1t   = (unsigned short*)take((size_t)2 * Hn * Hn * 2);
  unsigned short* w2t   = (unsigned short*)take((size_t)2 * Hn * Hn * 2);
  unsigned short* wkvt  = (unsigned short*)take((size_t)2 * Hn * Hn * 2);
  unsigned short* wqt   = (unsigned short*)take((size_t)Hn * Hn * 2);
  unsigned short* wopt  = (unsigned short*)take((size_t)Hn * Hn * 2);
  unsigned short* wot   = (unsigned short*)take((size_t)Vn * Hn * 2);
  float*          bkv   = (float*)take(2 * Hn * sizeof(float));
  unsigned short* hq    = (unsigned short*)take(Bn * Hn * 2);
  float*          qb    = (float*)take(Bn * Hn * sizeof(float));
  unsigned short* ctx16 = (unsigned short*)take(Bn * Hn * 2);
  unsigned short* t16   = (unsigned short*)take(Bn * Hn * 2);

  // ---- weight prep: transpose + bf16 convert (every call, capture-safe) ----
  dim3 tb(256);
  transpose_cvt_kernel<<<dim3(2 * Hn / 32, Hn / 32), tb, 0, stream>>>(ff_w1, w1t, Hn, 2 * Hn);
  transpose_cvt_kernel<<<dim3(Hn / 32, 2 * Hn / 32), tb, 0, stream>>>(ff_w2, w2t, 2 * Hn, Hn);
  for (int n = 0; n < NHn; ++n) {
    transpose_cvt_kernel<<<dim3(HHn / 32, Hn / 32), tb, 0, stream>>>(
        wk + (size_t)n * Hn * HHn, wkvt + (size_t)n * HHn * Hn, Hn, HHn);
    transpose_cvt_kernel<<<dim3(HHn / 32, Hn / 32), tb, 0, stream>>>(
        wv + (size_t)n * Hn * HHn, wkvt + (size_t)(Hn + n * HHn) * Hn, Hn, HHn);
    transpose_cvt_kernel<<<dim3(HHn / 32, Hn / 32), tb, 0, stream>>>(
        wq + (size_t)n * Hn * HHn, wqt + (size_t)n * HHn * Hn, Hn, HHn);
  }
  transpose_cvt_kernel<<<dim3(Hn / 32, Hn / 32), tb, 0, stream>>>(w_op, wopt, Hn, Hn);
  transpose_cvt_kernel<<<dim3(Vn / 32, Hn / 32), tb, 0, stream>>>(w_out, wot, Hn, Vn);
  hipMemcpyAsync(bkv, bk, Hn * sizeof(float), hipMemcpyDeviceToDevice, stream);
  hipMemcpyAsync(bkv + Hn, bv, Hn * sizeof(float), hipMemcpyDeviceToDevice, stream);

  // ---- pipeline ----
  gather_embed_kernel<<<Bn * Ln, 128, 0, stream>>>(seq, embed, e16);

  // f_pre = relu(e @ ff_w1 + b1)
  wmma_gemm_kernel<8, 1, true, true, false><<<dim3(Bn * Ln / 128, 2 * Hn / 64), 256, 0, stream>>>(
      e16, w1t, ff_b1, h1, Bn * Ln, 2 * Hn, Hn, nullptr, nullptr);
  // x = f_pre @ ff_w2 + b2 + e  (residual re-gathered from embed via seq)
  wmma_gemm_kernel<8, 1, false, false, true><<<dim3(Bn * Ln / 128, Hn / 64), 256, 0, stream>>>(
      h1, w2t, ff_b2, x32, Bn * Ln, Hn, 2 * Hn, seq, embed);
  layernorm_kernel<<<Bn * Ln, 128, 0, stream>>>(x32, ln_g, ln_b, h16);

  // [k|v] = h @ [wk|wv] + [bk|bv]  (one fused GEMM, fp32 out for the scan)
  wmma_gemm_kernel<8, 1, false, false, false><<<dim3(Bn * Ln / 128, 2 * Hn / 64), 256, 0, stream>>>(
      h16, wkvt, bkv, kv, Bn * Ln, 2 * Hn, Hn, nullptr, nullptr);

  gather_last_kernel<<<Bn, 256, 0, stream>>>(h16, hq);
  // q = h_last @ wq + bq   (skinny M=16 config: 1x8 waves)
  wmma_gemm_kernel<1, 8, false, false, false><<<dim3(1, 1), 256, 0, stream>>>(
      hq, wqt, bq, qb, Bn, Hn, Hn, nullptr, nullptr);

  // sequential delta-rule scan + ctx = M @ q (TDM-streamed k/v)
  delta_scan_kernel<<<Bn * NHn, 256, 0, stream>>>(kv, qb, ctx16);

  // t = ctx @ w_op + b_op
  wmma_gemm_kernel<1, 8, false, true, false><<<dim3(1, 1), 256, 0, stream>>>(
      ctx16, wopt, b_op, t16, Bn, Hn, Hn, nullptr, nullptr);
  // out = t @ w_out + b_out
  wmma_gemm_kernel<1, 8, false, false, false><<<dim3(1, (Vn + 511) / 512), 256, 0, stream>>>(
      t16, wot, b_out, out, Bn, Vn, Hn, nullptr, nullptr);
}
